// EdgeBlockSum_38345468018710
// MI455X (gfx1250) — matrix operand order repaired and checked by reading, not compiled, not run on hardware
//
#include <hip/hip_runtime.h>

// ---------------------------------------------------------------------------
// Fused GNN edge-block MLP for gfx1250 (MI455X), wave32 + WMMA f32_16x16x32_f16
// + async global->LDS weight staging (ASYNCcnt path)
// ---------------------------------------------------------------------------

typedef __attribute__((ext_vector_type(16))) _Float16 v16h;
typedef __attribute__((ext_vector_type(8)))  _Float16 v8h;
typedef __attribute__((ext_vector_type(8)))  float    v8f;
typedef __attribute__((ext_vector_type(4)))  float    v4f;
typedef int v4i __attribute__((vector_size(16)));   // matches builtin param type

#define DEVINL __device__ __forceinline__

#if __has_builtin(__builtin_amdgcn_global_load_async_to_lds_b128) && \
    __has_builtin(__builtin_amdgcn_s_wait_asynccnt)
#define USE_ASYNC_LDS 1
#else
#define USE_ASYNC_LDS 0
#endif

constexpr int HID  = 256;
constexpr int EDIM = 64;
// Padded row strides (halves): stride*2B must break the 64-bank pattern.
constexpr int LDW = 264;   // for 256-wide rows (W1, W2, h staging)
constexpr int LDE = 72;    // for 64-wide rows (edge_lin)

// ---- dynamic LDS layout for the edge kernel (bytes) ----
// The first SZ_WPAD bytes are an exact image of the pre-packed f16 weight blob,
// so staging is a flat 16B-chunk copy (async-to-LDS friendly).
constexpr size_t OFF_W1  = 0;
constexpr size_t SZ_W1   = 256 * LDW * 2;          // 135168
constexpr size_t OFF_EL  = OFF_W1 + SZ_W1;
constexpr size_t SZ_EL   = 256 * LDE * 2;          // 36864
constexpr size_t OFF_W2  = OFF_EL + SZ_EL;
constexpr size_t SZ_W2   = 64 * LDW * 2;           // 33792
constexpr size_t SZ_WPAD = SZ_W1 + SZ_EL + SZ_W2;  // 205824 (divisible by 16)
constexpr size_t OFF_B1  = SZ_WPAD;                // 256 f32
constexpr size_t OFF_B2  = OFF_B1 + 256 * 4;       // 64 f32
constexpr size_t OFF_G   = OFF_B2 + 64 * 4;        // 64 f32
constexpr size_t OFF_BE  = OFF_G  + 64 * 4;        // 64 f32
constexpr size_t OFF_STG = OFF_BE + 64 * 4;        // 8 waves * 16*LDW halves
constexpr size_t SZ_STG  = 8 * 16 * LDW * 2;       // 67584
constexpr size_t OFF_IDX = OFF_STG + SZ_STG;       // 8 waves * 32 ints
constexpr size_t SMEM_B  = OFF_IDX + 8 * 32 * 4;   // total = 276224 B (< 320KB WGP LDS)

DEVINL v8f wmma_f16(v16h a, v16h b, v8f c) {
  return __builtin_amdgcn_wmma_f32_16x16x32_f16(false, a, false, b,
                                                (short)0, c, false, false);
}

// A-matrix fragment (16x32 f16) from row-major f32 global memory.
// Per ISA 7.12.2: lanes 0-15 -> row=lane, K {0..7,16..23}; lanes 16-31 K {8..15,24..31}.
DEVINL v16h afrag_gf32(const float* __restrict__ base, long row0, int ld, int k0, int lane) {
  int m  = lane & 15;
  int ko = (lane >> 4) << 3;
  const float* p = base + (row0 + (long)m) * (long)ld + k0 + ko;
  v4f x0 = *(const v4f*)(p +  0);
  v4f x1 = *(const v4f*)(p +  4);
  v4f x2 = *(const v4f*)(p + 16);
  v4f x3 = *(const v4f*)(p + 20);
  v16h r;
#pragma unroll
  for (int i = 0; i < 4; ++i) {
    r[i]      = (_Float16)x0[i];
    r[4 + i]  = (_Float16)x1[i];
    r[8 + i]  = (_Float16)x2[i];
    r[12 + i] = (_Float16)x3[i];
  }
  return r;
}

// B-matrix fragment (32x16 f16) from row-major f32 weights (B col n == W row n).
// Layout (matches A): lane&15 = N, lanes 0-15 K {0..7,16..23}; lanes 16-31 K {8..15,24..31}.
DEVINL v16h bfrag_gf32(const float* __restrict__ base, long row0, int ld, int k0, int lane) {
  int n  = lane & 15;
  int kb = (lane >> 4) << 3;
  const float* p = base + (row0 + (long)n) * (long)ld + k0 + kb;
  v16h r;
#pragma unroll
  for (int j = 0; j < 4; ++j) {
    v4f x = *(const v4f*)(p + 4 * j + ((j >> 1) << 3));
#pragma unroll
    for (int i = 0; i < 4; ++i) r[4 * j + i] = (_Float16)x[i];
  }
  return r;
}

// A fragment from f16 LDS (row-major, padded stride)
DEVINL v16h afrag_lds(const _Float16* s, int ld, int k0, int lane) {
  int m  = lane & 15;
  int ko = (lane >> 4) << 3;
  const _Float16* p = s + m * ld + k0 + ko;
  v8h lo = *(const v8h*)p;
  v8h hi = *(const v8h*)(p + 16);
  v16h r;
#pragma unroll
  for (int i = 0; i < 8; ++i) { r[i] = lo[i]; r[8 + i] = hi[i]; }
  return r;
}

// B fragment from f16 LDS
DEVINL v16h bfrag_lds(const _Float16* s, int ld, int row0, int k0, int lane) {
  int n  = lane & 15;
  int kb = (lane >> 4) << 3;
  const _Float16* p = s + (row0 + n) * ld + k0 + kb;
  v8h lo = *(const v8h*)p;
  v8h hi = *(const v8h*)(p + 16);
  v16h r;
#pragma unroll
  for (int i = 0; i < 8; ++i) { r[i] = lo[i]; r[8 + i] = hi[i]; }
  return r;
}

// ---------------------------------------------------------------------------
// Kernel P: pack W1 / edge_lin / W2 into an f16 blob laid out exactly like the
// LDS image (padded row strides), so the edge kernel can flat-copy it.
// ---------------------------------------------------------------------------
__global__ void __launch_bounds__(256)
pack_weights_f16(const float* __restrict__ W1g, const float* __restrict__ ELg,
                 const float* __restrict__ W2g, _Float16* __restrict__ Wpad) {
  const int H1  = 256 * LDW;            // 67584 halves
  const int H2  = H1 + 256 * LDE;       // 86016
  const int TOT = H2 + 64 * LDW;        // 102912
  int i = blockIdx.x * 256 + threadIdx.x;
  if (i >= TOT) return;
  _Float16 v;
  if (i < H1) {
    int r = i / LDW, c = i % LDW;
    v = (c < 256) ? (_Float16)W1g[r * 256 + c] : (_Float16)0.0f;
  } else if (i < H2) {
    int j = i - H1;
    int r = j / LDE, c = j % LDE;
    v = (c < 64) ? (_Float16)ELg[r * 64 + c] : (_Float16)0.0f;
  } else {
    int j = i - H2;
    int r = j / LDW, c = j % LDW;
    v = (c < 256) ? (_Float16)W2g[r * 256 + c] : (_Float16)0.0f;
  }
  Wpad[i] = v;
}

// ---------------------------------------------------------------------------
// Kernel A: P_src = node @ src_lin^T ; P_dst = node @ dst_lin^T + bias  (f16 out)
// One wave per (16-row tile, 64-col group, src/dst).
// ---------------------------------------------------------------------------
__global__ void __launch_bounds__(256)
node_proj_wmma(const float* __restrict__ node,
               const float* __restrict__ Wsrc,
               const float* __restrict__ Wdst,
               const float* __restrict__ bias,
               _Float16* __restrict__ Psrc,
               _Float16* __restrict__ Pdst,
               int Nn) {
  const int tid  = threadIdx.x;
  const int lane = tid & 31;
  const int wave = tid >> 5;
  const int task = blockIdx.x * (blockDim.x >> 5) + wave;
  const int which = task & 1;          // 0=src, 1=dst
  const int cg    = (task >> 1) & 3;   // 64-col group
  const int rt    = task >> 3;         // 16-row tile
  const long r0   = (long)rt * 16;
  if (r0 + 16 > Nn) return;

  const float* W = which ? Wdst : Wsrc;
  const int n0   = cg * 64;

  v8f acc[4];
#pragma unroll
  for (int t = 0; t < 4; ++t)
#pragma unroll
    for (int v = 0; v < 8; ++v) acc[t][v] = 0.0f;

#pragma unroll
  for (int k0 = 0; k0 < HID; k0 += 32) {
    v16h a = afrag_gf32(node, r0, HID, k0, lane);
#pragma unroll
    for (int t = 0; t < 4; ++t) {
      v16h b = bfrag_gf32(W, n0 + 16 * t, HID, k0, lane);
      acc[t] = wmma_f16(a, b, acc[t]);
    }
  }

  _Float16* P = which ? Pdst : Psrc;
  const int n  = lane & 15;
  const int mh = (lane >> 4) << 3;
#pragma unroll
  for (int t = 0; t < 4; ++t)
#pragma unroll
    for (int v = 0; v < 8; ++v) {
      int col = n0 + 16 * t + n;
      float x = acc[t][v];
      if (which) x += bias[col];
      P[(size_t)(r0 + v + mh) * HID + col] = (_Float16)x;
    }
}

// ---------------------------------------------------------------------------
// Kernel B: fully fused per-edge pipeline. One wave = 16 edges.
//   h  = relu(edge@EL^T + Psrc[src] + Pdst[dst])
//   h2 = relu(h @ W1^T + b1)
//   y  = LN(h2 @ W2^T + b2)
// ---------------------------------------------------------------------------
__global__ void __launch_bounds__(256)
edge_mlp_wmma(const float* __restrict__ edge_attr,
              const long long* __restrict__ eidx,
              const _Float16* __restrict__ Wpad,
              const float* __restrict__ b1g,
              const float* __restrict__ b2g,
              const float* __restrict__ gammag,
              const float* __restrict__ betag,
              const _Float16* __restrict__ Psrc,
              const _Float16* __restrict__ Pdst,
              float* __restrict__ out,
              int E) {
  extern __shared__ char smem[];
  _Float16* sEL  = (_Float16*)(smem + OFF_EL);
  _Float16* sW1  = (_Float16*)(smem + OFF_W1);
  _Float16* sW2  = (_Float16*)(smem + OFF_W2);
  float*    sB1  = (float*)(smem + OFF_B1);
  float*    sB2  = (float*)(smem + OFF_B2);
  float*    sG   = (float*)(smem + OFF_G);
  float*    sBe  = (float*)(smem + OFF_BE);
  _Float16* sSTG = (_Float16*)(smem + OFF_STG);
  int*      sIDX = (int*)(smem + OFF_IDX);

  const int tid = threadIdx.x;

  // ---- stage pre-packed f16 weight blob into LDS (flat 16B-chunk copy) ----
  {
    constexpr int NCHUNK = (int)(SZ_WPAD / 16);   // 12864
    const char* g = (const char*)Wpad;
    char*       l = smem;
#if USE_ASYNC_LDS
    for (int i = tid; i < NCHUNK; i += 256) {
      __builtin_amdgcn_global_load_async_to_lds_b128(
          (v4i*)(g + (size_t)i * 16),
          (v4i*)(l + (size_t)i * 16),
          0, 0);
    }
#else
    for (int i = tid; i < NCHUNK; i += 256) {
      *(v8h*)(l + (size_t)i * 16) = *(const v8h*)(g + (size_t)i * 16);
    }
#endif
  }
  sB1[tid] = b1g[tid];
  if (tid < 64) { sB2[tid] = b2g[tid]; sG[tid] = gammag[tid]; sBe[tid] = betag[tid]; }
#if USE_ASYNC_LDS
  __builtin_amdgcn_s_wait_asynccnt(0);
#endif
  __syncthreads();

  const int lane = tid & 31;
  const int wave = tid >> 5;
  const long e0 = ((long)blockIdx.x * 8 + wave) * 16;
  if (e0 + 16 > E) return;

  _Float16* stg = sSTG + (size_t)wave * 16 * LDW;
  int* sS = sIDX + wave * 32;
  if (lane < 16) {
    sS[lane]      = (int)eidx[e0 + lane];            // src
    sS[16 + lane] = (int)eidx[(long)E + e0 + lane];  // dst
  }

  const int n  = lane & 15;
  const int mh = (lane >> 4) << 3;

  // Hoist per-row gather bases (row M = v + mh is fixed per lane-half).
  const _Float16* pS[8];
  const _Float16* pD[8];
#pragma unroll
  for (int v = 0; v < 8; ++v) {
    int si = sS[v + mh];
    int di = sS[16 + v + mh];
    pS[v] = Psrc + (size_t)si * HID + n;
    pD[v] = Pdst + (size_t)di * HID + n;
  }

  // ---- GEMM1: [16,64] x [64,256] via sEL, fused gather + relu -> staging ----
  v16h ea0 = afrag_gf32(edge_attr, e0, EDIM, 0, lane);
  v16h ea1 = afrag_gf32(edge_attr, e0, EDIM, 32, lane);
#pragma unroll
  for (int g = 0; g < 4; ++g) {
    v8f acc[4];
#pragma unroll
    for (int t = 0; t < 4; ++t)
#pragma unroll
      for (int v = 0; v < 8; ++v) acc[t][v] = 0.0f;
#pragma unroll
    for (int t = 0; t < 4; ++t) {
      v16h b0 = bfrag_lds(sEL, LDE, g * 64 + 16 * t, 0, lane);
      acc[t] = wmma_f16(ea0, b0, acc[t]);
      v16h b1f = bfrag_lds(sEL, LDE, g * 64 + 16 * t, 32, lane);
      acc[t] = wmma_f16(ea1, b1f, acc[t]);
    }
#pragma unroll
    for (int v = 0; v < 8; ++v) {
      int M = v + mh;
#pragma unroll
      for (int t = 0; t < 4; ++t) {
        int c = g * 64 + 16 * t;               // immediate offset off pS/pD bases
        float x = acc[t][v] + (float)pS[v][c] + (float)pD[v][c];
        x = fmaxf(x, 0.0f);
        stg[M * LDW + c + n] = (_Float16)x;
      }
    }
  }

  // ---- GEMM2: [16,256] x [256,256] via sW1, +b1, relu, restage ----
  v8f a2[16];
#pragma unroll
  for (int t = 0; t < 16; ++t)
#pragma unroll
    for (int v = 0; v < 8; ++v) a2[t][v] = 0.0f;
#pragma unroll
  for (int k = 0; k < 8; ++k) {
    v16h a = afrag_lds(stg, LDW, k * 32, lane);
#pragma unroll
    for (int t = 0; t < 16; ++t) {
      v16h b = bfrag_lds(sW1, LDW, 16 * t, k * 32, lane);
      a2[t] = wmma_f16(a, b, a2[t]);
    }
  }
#pragma unroll
  for (int t = 0; t < 16; ++t)
#pragma unroll
    for (int v = 0; v < 8; ++v) {
      int M = v + mh;
      int col = 16 * t + n;
      float x = fmaxf(a2[t][v] + sB1[col], 0.0f);
      stg[M * LDW + col] = (_Float16)x;   // all A reads above already consumed
    }

  // ---- GEMM3: [16,256] x [256,64] via sW2, +b2 ----
  v8f a3[4];
#pragma unroll
  for (int t = 0; t < 4; ++t)
#pragma unroll
    for (int v = 0; v < 8; ++v) a3[t][v] = 0.0f;
#pragma unroll
  for (int k = 0; k < 8; ++k) {
    v16h a = afrag_lds(stg, LDW, k * 32, lane);
#pragma unroll
    for (int t = 0; t < 4; ++t) {
      v16h b = bfrag_lds(sW2, LDW, 16 * t, k * 32, lane);
      a3[t] = wmma_f16(a, b, a3[t]);
    }
  }

  // ---- LayerNorm over 64 cols; row M lives in one 16-lane half-group ----
#pragma unroll
  for (int v = 0; v < 8; ++v) {
    float yv[4];
    float s = 0.0f, ss = 0.0f;
#pragma unroll
    for (int t = 0; t < 4; ++t) {
      yv[t] = a3[t][v] + sB2[16 * t + n];
      s  += yv[t];
      ss += yv[t] * yv[t];
    }
#pragma unroll
    for (int m = 1; m < 16; m <<= 1) {
      s  += __shfl_xor(s, m, 32);
      ss += __shfl_xor(ss, m, 32);
    }
    float mu  = s * (1.0f / 64.0f);
    float var = ss * (1.0f / 64.0f) - mu * mu;
    float rs  = rsqrtf(var + 1e-5f);
    long row = e0 + v + mh;
#pragma unroll
    for (int t = 0; t < 4; ++t) {
      int col = 16 * t + n;
      out[row * 64 + col] = (yv[t] - mu) * rs * sG[col] + sBe[col];
    }
  }
}

// ---------------------------------------------------------------------------
extern "C" void kernel_launch(void* const* d_in, const int* in_sizes, int n_in,
                              void* d_out, int out_size, void* d_ws, size_t ws_size,
                              hipStream_t stream) {
  const float*     edge_attr = (const float*)d_in[0];
  const float*     node_attr = (const float*)d_in[1];
  const long long* eidx      = (const long long*)d_in[2];
  const float*     edge_lin  = (const float*)d_in[3];
  const float*     src_lin   = (const float*)d_in[4];
  const float*     dst_lin   = (const float*)d_in[5];
  const float*     bias      = (const float*)d_in[6];
  const float*     W1        = (const float*)d_in[7];
  const float*     b1        = (const float*)d_in[8];
  const float*     W2        = (const float*)d_in[9];
  const float*     b2        = (const float*)d_in[10];
  const float*     gamma     = (const float*)d_in[11];
  const float*     beta      = (const float*)d_in[12];
  float* out = (float*)d_out;

  const int Nn = in_sizes[1] / HID;   // 50000
  const int E  = in_sizes[2] / 2;     // 800000

  _Float16* Psrc = (_Float16*)d_ws;
  _Float16* Pdst = Psrc + (size_t)Nn * HID;
  _Float16* Wpad = Pdst + (size_t)Nn * HID;   // byte offset Nn*1024, 16B-aligned

  // Kernel P: pack/pad weights to f16 LDS image (102912 halves)
  {
    const int TOT = 256 * LDW + 256 * LDE + 64 * LDW;
    pack_weights_f16<<<(TOT + 255) / 256, 256, 0, stream>>>(W1, edge_lin, W2, Wpad);
  }

  // Kernel A: tasks = rowTiles * 4 colgroups * 2 (src/dst); 8 waves per block
  const int rowTiles = (Nn + 15) / 16;
  const int blocksA  = rowTiles;  // rowTiles*8 tasks / 8 waves per block
  node_proj_wmma<<<blocksA, 256, 0, stream>>>(node_attr, src_lin, dst_lin, bias,
                                              Psrc, Pdst, Nn);

  // Kernel B: one wave per 16 edges, 8 waves per block
  const int eTiles  = (E + 15) / 8;
  const int blocksB = ((E + 15) / 16 + 7) / 8;
  edge_mlp_wmma<<<blocksB, 256, SMEM_B, stream>>>(edge_attr, eidx, Wpad,
                                                  b1, b2, gamma, beta,
                                                  Psrc, Pdst, out, E);
}
